// Decoder_63780264346225
// MI455X (gfx1250) — matrix operand order repaired and checked by reading, not compiled, not run on hardware
//
#include <hip/hip_runtime.h>
#include <hip/hip_bf16.h>
#include <math.h>

// ---------------- problem constants ----------------
#define BN_   650
#define TT    48
#define CC    128
#define VV    3
#define T2_   12
#define LL    4
#define HH    8
#define DD    64
#define HD_   512
#define NB_   6
#define NHID_ 4
#define DIN_  387
#define DINP  416          // 387 padded to mult of 32
#define CY_   515
#define CYP   544          // 515 padded to mult of 32
#define M1    (BN_*TT)     // 31200
#define M2    (BN_*T2_)    // 7800
#define LOG2PI_ 1.8378770664093453f
#define CDIV(a,b) (((a)+(b)-1)/(b))

typedef __attribute__((ext_vector_type(16))) __bf16 v16bf;
typedef __attribute__((ext_vector_type(8)))  float  v8f;

union BFr { v16bf v; uint4 q[2]; };

__device__ __forceinline__ __bf16 f2bf(float x) { return (__bf16)x; }

// ---------------- WMMA GEMM: C[M,N] = A[M,K] * Bt[N,K]^T + bias ----------------
// A: bf16 row-major, leading dim lda (multiple of 32, zero-padded K tail)
// Bt: bf16, row n holds column n of B over K (leading dim lda)
// One wave -> 16x64 tile (4x v_wmma_f32_16x16x32_bf16 sharing the A fragment)
// Block: 8 waves -> 64(M) x 128(N)
__global__ __launch_bounds__(256)
void gemm_bf16(const __bf16* __restrict__ A, const __bf16* __restrict__ Bt,
               const float* __restrict__ bias, float* __restrict__ C,
               int M, int N, int Kp, int lda, int ldc)
{
    const int lane = threadIdx.x & 31;
    const int wave = threadIdx.x >> 5;
    const int wm = wave & 3, wn = wave >> 2;
    const int rowBase = blockIdx.y * 64 + wm * 16;
    const int colBase = blockIdx.x * 128 + wn * 64;
    const int hl = lane >> 4;          // lane half
    const int r  = lane & 15;

    v8f acc0 = {}, acc1 = {}, acc2 = {}, acc3 = {};

    int arow = rowBase + r; if (arow > M - 1) arow = M - 1;
    const __bf16* aptr = A + (size_t)arow * lda + hl * 16;

    int bn0 = colBase      + r; if (bn0 > N - 1) bn0 = N - 1;
    int bn1 = colBase + 16 + r; if (bn1 > N - 1) bn1 = N - 1;
    int bn2 = colBase + 32 + r; if (bn2 > N - 1) bn2 = N - 1;
    int bn3 = colBase + 48 + r; if (bn3 > N - 1) bn3 = N - 1;
    const __bf16* bp0 = Bt + (size_t)bn0 * lda + hl * 16;
    const __bf16* bp1 = Bt + (size_t)bn1 * lda + hl * 16;
    const __bf16* bp2 = Bt + (size_t)bn2 * lda + hl * 16;
    const __bf16* bp3 = Bt + (size_t)bn3 * lda + hl * 16;

    for (int kk = 0; kk < Kp; kk += 32) {
        BFr a;
        a.q[0] = *reinterpret_cast<const uint4*>(aptr + kk);       // K = h*16 .. h*16+7
        a.q[1] = *reinterpret_cast<const uint4*>(aptr + kk + 8);   // K = h*16+8 .. h*16+15
        BFr b0, b1, b2, b3;
        b0.q[0] = *reinterpret_cast<const uint4*>(bp0 + kk);
        b0.q[1] = *reinterpret_cast<const uint4*>(bp0 + kk + 8);
        b1.q[0] = *reinterpret_cast<const uint4*>(bp1 + kk);
        b1.q[1] = *reinterpret_cast<const uint4*>(bp1 + kk + 8);
        b2.q[0] = *reinterpret_cast<const uint4*>(bp2 + kk);
        b2.q[1] = *reinterpret_cast<const uint4*>(bp2 + kk + 8);
        b3.q[0] = *reinterpret_cast<const uint4*>(bp3 + kk);
        b3.q[1] = *reinterpret_cast<const uint4*>(bp3 + kk + 8);
        acc0 = __builtin_amdgcn_wmma_f32_16x16x32_bf16(false, a.v, false, b0.v, (short)0, acc0, false, false);
        acc1 = __builtin_amdgcn_wmma_f32_16x16x32_bf16(false, a.v, false, b1.v, (short)0, acc1, false, false);
        acc2 = __builtin_amdgcn_wmma_f32_16x16x32_bf16(false, a.v, false, b2.v, (short)0, acc2, false, false);
        acc3 = __builtin_amdgcn_wmma_f32_16x16x32_bf16(false, a.v, false, b3.v, (short)0, acc3, false, false);
    }

    // C/D layout: VGPR i -> rows rowBase+i (lanes 0-15) / rowBase+8+i (lanes 16-31), col = lane&15
    const int col = colBase + r;
    const int rtop = rowBase + hl * 8;
    for (int i = 0; i < 8; ++i) {
        int row = rtop + i;
        if (row < M) {
            size_t base = (size_t)row * ldc;
            if (col < N)      C[base + col]      = acc0[i] + (bias ? bias[col]      : 0.f);
            if (col + 16 < N) C[base + col + 16] = acc1[i] + (bias ? bias[col + 16] : 0.f);
            if (col + 32 < N) C[base + col + 32] = acc2[i] + (bias ? bias[col + 32] : 0.f);
            if (col + 48 < N) C[base + col + 48] = acc3[i] + (bias ? bias[col + 48] : 0.f);
        }
    }
}

// ---------------- data prep kernels ----------------
// evA[bn*48+t][col] = col<387 ? (j<128? enc[bn,v,t,j] : tv[bn,v,t]) : 0  (col = v*129+j)
__global__ void build_ev(const float* __restrict__ enc, const float* __restrict__ tv,
                         __bf16* __restrict__ evA)
{
    size_t idx = (size_t)blockIdx.x * blockDim.x + threadIdx.x;
    size_t tot = (size_t)M1 * DINP;
    if (idx >= tot) return;
    int col = (int)(idx % DINP);
    size_t row = idx / DINP;
    int bn = (int)(row / TT), t = (int)(row % TT);
    float val = 0.f;
    if (col < DIN_) {
        int v = col / (CC + 1), j = col % (CC + 1);
        if (j < CC) val = enc[(((size_t)bn * VV + v) * TT + t) * CC + j];
        else        val = tv[((size_t)bn * VV + v) * TT + t];
    }
    evA[idx] = f2bf(val);
}

// attA[bn*12+t2][col] = enc[bn,v,36+t2,j]   (col = v*128+j), ld = 384
__global__ void build_attin(const float* __restrict__ enc, __bf16* __restrict__ attA)
{
    size_t idx = (size_t)blockIdx.x * blockDim.x + threadIdx.x;
    size_t tot = (size_t)M2 * (VV * CC);
    if (idx >= tot) return;
    int col = (int)(idx % (VV * CC));
    size_t row = idx / (VV * CC);
    int bn = (int)(row / T2_), t2 = (int)(row % T2_);
    int v = col / CC, j = col % CC;
    attA[idx] = f2bf(enc[(((size_t)bn * VV + v) * TT + (TT - T2_ + t2)) * CC + j]);
}

// dst[b][n][k] = src[b][k][n] (k<K else 0), dst row stride ldk
__global__ void transpose_w(const float* __restrict__ src, __bf16* __restrict__ dst,
                            int batch, int K, int N, int ldk)
{
    size_t idx = (size_t)blockIdx.x * blockDim.x + threadIdx.x;
    size_t tot = (size_t)batch * N * ldk;
    if (idx >= tot) return;
    int k = (int)(idx % ldk);
    size_t rem = idx / ldk;
    int n = (int)(rem % N);
    int b = (int)(rem / N);
    float v = (k < K) ? src[((size_t)b * K + k) * N + n] : 0.f;
    dst[idx] = f2bf(v);
}

// dst[m][c] = act(src[m*lds+c]) for c<K else 0 ; act 0=none 1=relu 2=tanh
__global__ void convert_act(const float* __restrict__ src, __bf16* __restrict__ dst,
                            int Mr, int K, int lds, int ldd, int act)
{
    size_t idx = (size_t)blockIdx.x * blockDim.x + threadIdx.x;
    size_t tot = (size_t)Mr * ldd;
    if (idx >= tot) return;
    int c = (int)(idx % ldd);
    size_t m = idx / ldd;
    float v = 0.f;
    if (c < K) {
        v = src[m * lds + c];
        if (act == 1) v = fmaxf(v, 0.f);
        else if (act == 2) v = tanhf(v);
    }
    dst[idx] = f2bf(v);
}

// ---------------- attention (per bn,head) ----------------
__global__ __launch_bounds__(64)
void attention_k(const float* __restrict__ attv, const float* __restrict__ keys,
                 const float* __restrict__ vals, float* __restrict__ att)
{
    int bn = blockIdx.x / HH;
    int h  = blockIdx.x % HH;
    __shared__ float q[T2_][DD];
    __shared__ float sc[T2_][TT];
    int t = threadIdx.x;
    for (int idx = t; idx < T2_ * DD; idx += 64) {
        int v = idx / DD, d = idx % DD;
        q[v][d] = attv[((size_t)bn * T2_ + v) * HD_ + h * DD + d];
    }
    __syncthreads();
    const float scale = 0.125f;  // D^-0.5
    if (t < TT) {
        const float* krow = keys + ((size_t)bn * TT + t) * HD_ + h * DD;
        for (int v = 0; v < T2_; ++v) {
            if (t >= (TT - T2_) + v) { sc[v][t] = -3.0e38f; }
            else {
                float s = 0.f;
                for (int d = 0; d < DD; ++d) s += q[v][d] * krow[d];
                sc[v][t] = s * scale;
            }
        }
    }
    __syncthreads();
    if (t < T2_) {
        float mx = -3.0e38f;
        for (int w = 0; w < TT; ++w) mx = fmaxf(mx, sc[t][w]);
        float sum = 0.f;
        for (int w = 0; w < TT; ++w) { float e = expf(sc[t][w] - mx); sc[t][w] = e; sum += e; }
        float inv = 1.f / sum;
        for (int w = 0; w < TT; ++w) sc[t][w] *= inv;
    }
    __syncthreads();
    for (int idx = t; idx < T2_ * DD; idx += 64) {
        int v = idx / DD, d = idx % DD;
        float s = 0.f;
        for (int w = 0; w < TT; ++w)
            s += sc[v][w] * vals[((size_t)bn * TT + w) * HD_ + h * DD + d];
        att[((size_t)bn * T2_ + v) * HD_ + h * DD + d] = s;
    }
}

// ---------------- residual + LayerNorm (row of 512) ----------------
__global__ __launch_bounds__(256)
void ln_residual(const float* __restrict__ x, const float* __restrict__ res,
                 const float* __restrict__ s, const float* __restrict__ b,
                 float* __restrict__ outF, __bf16* __restrict__ outB)
{
    int row = blockIdx.x;
    __shared__ float r1[256], r2[256];
    int t = threadIdx.x;
    size_t base = (size_t)row * HD_;
    float v0 = x[base + t]       + res[base + t];
    float v1 = x[base + t + 256] + res[base + t + 256];
    r1[t] = v0 + v1;
    r2[t] = v0 * v0 + v1 * v1;
    __syncthreads();
    for (int o = 128; o > 0; o >>= 1) {
        if (t < o) { r1[t] += r1[t + o]; r2[t] += r2[t + o]; }
        __syncthreads();
    }
    float m = r1[0] * (1.f / HD_);
    float var = r2[0] * (1.f / HD_) - m * m;
    float rinv = rsqrtf(var + 1e-5f);
    float o0 = (v0 - m) * rinv * s[t]       + b[t];
    float o1 = (v1 - m) * rinv * s[t + 256] + b[t + 256];
    outF[base + t] = o0;
    outF[base + t + 256] = o1;
    if (outB) { outB[base + t] = f2bf(o0); outB[base + t + 256] = f2bf(o1); }
}

// ---------------- RealNVP helpers ----------------
__global__ void build_inp(const __bf16* __restrict__ y, __bf16* __restrict__ inpA)
{
    size_t idx = (size_t)blockIdx.x * blockDim.x + threadIdx.x;
    size_t tot = (size_t)M2 * CYP;
    if (idx >= tot) return;
    int c = (int)(idx % CYP);
    size_t row = idx / CYP;
    inpA[idx] = (c < HD_) ? y[row * HD_ + c] : f2bf(0.f);
}

__global__ void init_u(const float* __restrict__ tv, float* __restrict__ u,
                       float* __restrict__ logdet)
{
    int idx = blockIdx.x * blockDim.x + threadIdx.x;
    if (idx >= M2 * VV) return;
    int row = idx / VV, v = idx % VV;
    int bn = row / T2_, t2 = row % T2_;
    u[idx] = tv[((size_t)bn * VV + v) * TT + (TT - T2_ + t2)];
    logdet[idx] = 0.f;
}

__global__ void set_mu(const float* __restrict__ u, __bf16* __restrict__ inpA, int blk)
{
    int idx = blockIdx.x * blockDim.x + threadIdx.x;
    if (idx >= M2 * VV) return;
    int row = idx / VV, v = idx % VV;
    float mask = (float)((v + blk) & 1);
    inpA[(size_t)row * CYP + HD_ + v] = f2bf(mask * u[idx]);
}

__global__ void coupling_k(float* __restrict__ u, float* __restrict__ logdet,
                           const float* __restrict__ sF, const float* __restrict__ tF, int blk)
{
    int idx = blockIdx.x * blockDim.x + threadIdx.x;
    if (idx >= M2 * VV) return;
    int v = idx % VV;
    float mask = (float)((v + blk) & 1);
    float uu = u[idx], s = sF[idx], t = tF[idx];
    float mu = uu * mask;
    u[idx] = mu + (1.f - mask) * (uu - t) * expf(-s);
    logdet[idx] -= (1.f - mask) * s;
}

__global__ __launch_bounds__(256)
void bn_stats(const float* __restrict__ u, float* __restrict__ bm, float* __restrict__ bv)
{
    int p = blockIdx.x;  // t2*3+v  (36 positions)
    __shared__ float r1[256], r2[256];
    int t = threadIdx.x;
    float s1 = 0.f, s2 = 0.f;
    for (int bn = t; bn < BN_; bn += 256) {
        float x = u[(size_t)bn * (T2_ * VV) + p];
        s1 += x; s2 += x * x;
    }
    r1[t] = s1; r2[t] = s2;
    __syncthreads();
    for (int o = 128; o > 0; o >>= 1) {
        if (t < o) { r1[t] += r1[t + o]; r2[t] += r2[t + o]; }
        __syncthreads();
    }
    if (t == 0) {
        float m = r1[0] * (1.f / BN_);
        bm[p] = m;
        bv[p] = r2[0] * (1.f / BN_) - m * m;
    }
}

__global__ void bn_apply(float* __restrict__ u, float* __restrict__ logdet,
                         const float* __restrict__ bm, const float* __restrict__ bv,
                         const float* __restrict__ lg, const float* __restrict__ beta)
{
    int idx = blockIdx.x * blockDim.x + threadIdx.x;
    if (idx >= M2 * VV) return;
    int p = idx % (T2_ * VV);
    int v = idx % VV;
    float m = bm[p], var = bv[p];
    float g = lg[v], bt = beta[v];
    float x = u[idx];
    u[idx] = expf(g) * (x - m) * rsqrtf(var + 1e-5f) + bt;
    logdet[idx] += g - 0.5f * logf(var + 1e-5f);
}

__global__ __launch_bounds__(64)
void final_reduce(const float* __restrict__ u, const float* __restrict__ logdet,
                  float* __restrict__ out)
{
    int bn = blockIdx.x;
    __shared__ float red[64];
    int t = threadIdx.x;
    float s = 0.f;
    for (int i = t; i < T2_ * VV; i += 64) {
        float uu = u[(size_t)bn * (T2_ * VV) + i];
        float ld = logdet[(size_t)bn * (T2_ * VV) + i];
        s += 0.5f * uu * uu + 0.5f * LOG2PI_ - ld;
    }
    red[t] = s;
    __syncthreads();
    for (int o = 32; o > 0; o >>= 1) { if (t < o) red[t] += red[t + o]; __syncthreads(); }
    if (t == 0) out[bn] = red[0];
}

// ---------------- host orchestration ----------------
extern "C" void kernel_launch(void* const* d_in, const int* in_sizes, int n_in,
                              void* d_out, int out_size, void* d_ws, size_t ws_size,
                              hipStream_t stream)
{
    const float* encoded = (const float*)d_in[0];
    const float* true_v  = (const float*)d_in[1];
    const float* W_shift = (const float*)d_in[2];
    const float* b_shift = (const float*)d_in[3];
    const float* W_key   = (const float*)d_in[4];
    const float* b_key   = (const float*)d_in[5];
    const float* W_val   = (const float*)d_in[6];
    const float* b_val   = (const float*)d_in[7];
    const float* ln1_s   = (const float*)d_in[8];
    const float* ln1_b   = (const float*)d_in[9];
    const float* ff_w1   = (const float*)d_in[10];
    const float* ff_b1   = (const float*)d_in[11];
    const float* ff_w2   = (const float*)d_in[12];
    const float* ff_b2   = (const float*)d_in[13];
    const float* ln2_s   = (const float*)d_in[14];
    const float* ln2_b   = (const float*)d_in[15];
    const float* s_w_in  = (const float*)d_in[16];
    const float* s_b_in  = (const float*)d_in[17];
    const float* s_w_hid = (const float*)d_in[18];
    const float* s_b_hid = (const float*)d_in[19];
    const float* s_w_out = (const float*)d_in[20];
    const float* s_b_out = (const float*)d_in[21];
    const float* t_w_in  = (const float*)d_in[22];
    const float* t_b_in  = (const float*)d_in[23];
    const float* t_w_hid = (const float*)d_in[24];
    const float* t_b_hid = (const float*)d_in[25];
    const float* t_w_out = (const float*)d_in[26];
    const float* t_b_out = (const float*)d_in[27];
    const float* bn_lg   = (const float*)d_in[28];
    const float* bn_bt   = (const float*)d_in[29];

    char* w = (char*)d_ws;
    auto alloc = [&](size_t bytes) -> void* {
        void* p = (void*)w;
        w += (bytes + 255) & ~(size_t)255;
        return p;
    };
    __bf16* evA    = (__bf16*)alloc((size_t)M1 * DINP * 2);
    __bf16* attA   = (__bf16*)alloc((size_t)M2 * VV * CC * 2);
    __bf16* BtK    = (__bf16*)alloc((size_t)LL * HD_ * DINP * 2);
    __bf16* BtV    = (__bf16*)alloc((size_t)LL * HD_ * DINP * 2);
    __bf16* BtSh   = (__bf16*)alloc((size_t)HD_ * (VV * CC) * 2);
    __bf16* Btff1  = (__bf16*)alloc((size_t)LL * HD_ * HD_ * 2);
    __bf16* Btff2  = (__bf16*)alloc((size_t)LL * HD_ * HD_ * 2);
    __bf16* BtSin  = (__bf16*)alloc((size_t)NB_ * HD_ * CYP * 2);
    __bf16* BtTin  = (__bf16*)alloc((size_t)NB_ * HD_ * CYP * 2);
    __bf16* BtShid = (__bf16*)alloc((size_t)NB_ * NHID_ * HD_ * HD_ * 2);
    __bf16* BtThid = (__bf16*)alloc((size_t)NB_ * NHID_ * HD_ * HD_ * 2);
    __bf16* BtSout = (__bf16*)alloc((size_t)NB_ * VV * HD_ * 2);
    __bf16* BtTout = (__bf16*)alloc((size_t)NB_ * VV * HD_ * 2);
    float*  keysF  = (float*)alloc((size_t)M1 * HD_ * 4);
    float*  valsF  = (float*)alloc((size_t)M1 * HD_ * 4);
    float*  attvF  = (float*)alloc((size_t)M2 * HD_ * 4);
    __bf16* attvA  = (__bf16*)alloc((size_t)M2 * HD_ * 2);
    float*  attF   = (float*)alloc((size_t)M2 * HD_ * 4);
    float*  tmpF   = (float*)alloc((size_t)M2 * HD_ * 4);
    __bf16* tmpA   = (__bf16*)alloc((size_t)M2 * HD_ * 2);
    __bf16* inpA   = (__bf16*)alloc((size_t)M2 * CYP * 2);
    float*  sF     = (float*)alloc((size_t)M2 * VV * 4);
    float*  tF     = (float*)alloc((size_t)M2 * VV * 4);
    float*  uB     = (float*)alloc((size_t)M2 * VV * 4);
    float*  ldB    = (float*)alloc((size_t)M2 * VV * 4);
    float*  bmB    = (float*)alloc(36 * 4);
    float*  bvB    = (float*)alloc(36 * 4);

    const int TB = 256;
    // ---- prep: activations & transposed bf16 weights ----
    build_ev<<<CDIV((size_t)M1 * DINP, TB), TB, 0, stream>>>(encoded, true_v, evA);
    build_attin<<<CDIV((size_t)M2 * VV * CC, TB), TB, 0, stream>>>(encoded, attA);
    transpose_w<<<CDIV((size_t)LL * HH * DD * DINP, TB), TB, 0, stream>>>(W_key, BtK, LL * HH, DIN_, DD, DINP);
    transpose_w<<<CDIV((size_t)LL * HH * DD * DINP, TB), TB, 0, stream>>>(W_val, BtV, LL * HH, DIN_, DD, DINP);
    transpose_w<<<CDIV((size_t)HD_ * VV * CC, TB), TB, 0, stream>>>(W_shift, BtSh, 1, VV * CC, HD_, VV * CC);
    transpose_w<<<CDIV((size_t)LL * HD_ * HD_, TB), TB, 0, stream>>>(ff_w1, Btff1, LL, HD_, HD_, HD_);
    transpose_w<<<CDIV((size_t)LL * HD_ * HD_, TB), TB, 0, stream>>>(ff_w2, Btff2, LL, HD_, HD_, HD_);
    transpose_w<<<CDIV((size_t)NB_ * HD_ * CYP, TB), TB, 0, stream>>>(s_w_in, BtSin, NB_, CY_, HD_, CYP);
    transpose_w<<<CDIV((size_t)NB_ * HD_ * CYP, TB), TB, 0, stream>>>(t_w_in, BtTin, NB_, CY_, HD_, CYP);
    transpose_w<<<CDIV((size_t)NB_ * NHID_ * HD_ * HD_, TB), TB, 0, stream>>>(s_w_hid, BtShid, NB_ * NHID_, HD_, HD_, HD_);
    transpose_w<<<CDIV((size_t)NB_ * NHID_ * HD_ * HD_, TB), TB, 0, stream>>>(t_w_hid, BtThid, NB_ * NHID_, HD_, HD_, HD_);
    transpose_w<<<CDIV((size_t)NB_ * VV * HD_, TB), TB, 0, stream>>>(s_w_out, BtSout, NB_, HD_, VV, HD_);
    transpose_w<<<CDIV((size_t)NB_ * VV * HD_, TB), TB, 0, stream>>>(t_w_out, BtTout, NB_, HD_, VV, HD_);
    init_u<<<CDIV(M2 * VV, TB), TB, 0, stream>>>(true_v, uB, ldB);

    // ---- initial queries: att_value = att_in @ W_shift + b ----
    gemm_bf16<<<dim3(CDIV(HD_, 128), CDIV(M2, 64)), 256, 0, stream>>>(
        attA, BtSh, b_shift, attvF, M2, HD_, VV * CC, VV * CC, HD_);

    // ---- attention layers ----
    for (int l = 0; l < LL; ++l) {
        gemm_bf16<<<dim3(CDIV(HD_, 128), CDIV(M1, 64)), 256, 0, stream>>>(
            evA, BtK + (size_t)l * HD_ * DINP, b_key + l * HD_, keysF, M1, HD_, DINP, DINP, HD_);
        gemm_bf16<<<dim3(CDIV(HD_, 128), CDIV(M1, 64)), 256, 0, stream>>>(
            evA, BtV + (size_t)l * HD_ * DINP, b_val + l * HD_, valsF, M1, HD_, DINP, DINP, HD_);
        attention_k<<<BN_ * HH, 64, 0, stream>>>(attvF, keysF, valsF, attF);
        ln_residual<<<M2, 256, 0, stream>>>(attF, attvF, ln1_s + l * HD_, ln1_b + l * HD_, attvF, attvA);
        gemm_bf16<<<dim3(CDIV(HD_, 128), CDIV(M2, 64)), 256, 0, stream>>>(
            attvA, Btff1 + (size_t)l * HD_ * HD_, ff_b1 + l * HD_, tmpF, M2, HD_, HD_, HD_, HD_);
        convert_act<<<CDIV((size_t)M2 * HD_, TB), TB, 0, stream>>>(tmpF, tmpA, M2, HD_, HD_, HD_, 1);
        gemm_bf16<<<dim3(CDIV(HD_, 128), CDIV(M2, 64)), 256, 0, stream>>>(
            tmpA, Btff2 + (size_t)l * HD_ * HD_, ff_b2 + l * HD_, attF, M2, HD_, HD_, HD_, HD_);
        ln_residual<<<M2, 256, 0, stream>>>(attF, attvF, ln2_s + l * HD_, ln2_b + l * HD_, attvF, attvA);
    }

    // ---- RealNVP flow ----
    build_inp<<<CDIV((size_t)M2 * CYP, TB), TB, 0, stream>>>(attvA, inpA);
    for (int blk = 0; blk < NB_; ++blk) {
        set_mu<<<CDIV(M2 * VV, TB), TB, 0, stream>>>(uB, inpA, blk);
        // s-net (tanh)
        gemm_bf16<<<dim3(CDIV(HD_, 128), CDIV(M2, 64)), 256, 0, stream>>>(
            inpA, BtSin + (size_t)blk * HD_ * CYP, s_b_in + blk * HD_, tmpF, M2, HD_, CYP, CYP, HD_);
        for (int i = 0; i < NHID_; ++i) {
            convert_act<<<CDIV((size_t)M2 * HD_, TB), TB, 0, stream>>>(tmpF, tmpA, M2, HD_, HD_, HD_, 2);
            gemm_bf16<<<dim3(CDIV(HD_, 128), CDIV(M2, 64)), 256, 0, stream>>>(
                tmpA, BtShid + (size_t)(blk * NHID_ + i) * HD_ * HD_,
                s_b_hid + (blk * NHID_ + i) * HD_, tmpF, M2, HD_, HD_, HD_, HD_);
        }
        convert_act<<<CDIV((size_t)M2 * HD_, TB), TB, 0, stream>>>(tmpF, tmpA, M2, HD_, HD_, HD_, 2);
        gemm_bf16<<<dim3(1, CDIV(M2, 64)), 256, 0, stream>>>(
            tmpA, BtSout + (size_t)blk * VV * HD_, s_b_out + blk * VV, sF, M2, VV, HD_, HD_, VV);
        // t-net (relu)
        gemm_bf16<<<dim3(CDIV(HD_, 128), CDIV(M2, 64)), 256, 0, stream>>>(
            inpA, BtTin + (size_t)blk * HD_ * CYP, t_b_in + blk * HD_, tmpF, M2, HD_, CYP, CYP, HD_);
        for (int i = 0; i < NHID_; ++i) {
            convert_act<<<CDIV((size_t)M2 * HD_, TB), TB, 0, stream>>>(tmpF, tmpA, M2, HD_, HD_, HD_, 1);
            gemm_bf16<<<dim3(CDIV(HD_, 128), CDIV(M2, 64)), 256, 0, stream>>>(
                tmpA, BtThid + (size_t)(blk * NHID_ + i) * HD_ * HD_,
                t_b_hid + (blk * NHID_ + i) * HD_, tmpF, M2, HD_, HD_, HD_, HD_);
        }
        convert_act<<<CDIV((size_t)M2 * HD_, TB), TB, 0, stream>>>(tmpF, tmpA, M2, HD_, HD_, HD_, 1);
        gemm_bf16<<<dim3(1, CDIV(M2, 64)), 256, 0, stream>>>(
            tmpA, BtTout + (size_t)blk * VV * HD_, t_b_out + blk * VV, tF, M2, VV, HD_, HD_, VV);
        // coupling + batch-norm flow
        coupling_k<<<CDIV(M2 * VV, TB), TB, 0, stream>>>(uB, ldB, sF, tF, blk);
        bn_stats<<<T2_ * VV, 256, 0, stream>>>(uB, bmB, bvB);
        bn_apply<<<CDIV(M2 * VV, TB), TB, 0, stream>>>(uB, ldB, bmB, bvB, bn_lg + blk * VV, bn_bt + blk * VV);
    }

    final_reduce<<<BN_, 64, 0, stream>>>(uB, ldB, (float*)d_out);
    (void)in_sizes; (void)n_in; (void)out_size; (void)ws_size;
}
